// GraphAttention_71837622993325
// MI455X (gfx1250) — matrix operand order repaired and checked by reading, not compiled, not run on hardware
//
#include <hip/hip_runtime.h>
#include <hip/hip_bf16.h>
#include <math.h>

#define N_NODES 4096
#define F_INN   512
#define F_OUTT  64
#define HEADS   8
#define ALPHAF  0.2f

typedef __attribute__((ext_vector_type(16))) _Float16 v16h;
typedef __attribute__((ext_vector_type(8)))  _Float16 v8h;
typedef __attribute__((ext_vector_type(8)))  float    v8f;
typedef __attribute__((ext_vector_type(4)))  float    v4f;
typedef __attribute__((ext_vector_type(4)))  int      v4i;

// ---------------- conversion kernels ----------------

__global__ void cvt_x_kernel(const float* __restrict__ x, _Float16* __restrict__ xb) {
    int idx = blockIdx.x * 256 + threadIdx.x;            // N*F_IN threads
    xb[idx] = (_Float16)x[idx];
}

// W [H][F_IN][F_OUT] f32 -> WbT [H][F_OUT][F_IN] f16 (transposed per head)
__global__ void cvt_w_kernel(const float* __restrict__ W, _Float16* __restrict__ WbT) {
    int idx = blockIdx.x * 256 + threadIdx.x;            // H*F_IN*F_OUT threads
    int h   = idx / (F_INN * F_OUTT);
    int rem = idx % (F_INN * F_OUTT);
    int kk  = rem / F_OUTT;
    int o   = rem % F_OUTT;
    WbT[((size_t)h * F_OUTT + o) * F_INN + kk] = (_Float16)W[idx];
}

// ---------------- Wh = x @ W (per head), WMMA f16, output transposed ----------------
// grid = (N/16)*HEADS blocks of 1 wave (32 threads). Wave computes 16x64 tile.
__global__ void wh_gemm_kernel(const _Float16* __restrict__ xb,
                               const _Float16* __restrict__ WbT,
                               _Float16* __restrict__ WhT) {
    const int bx   = blockIdx.x;
    const int h    = bx % HEADS;
    const int i0   = (bx / HEADS) * 16;
    const int lane = threadIdx.x & 31;
    const int lo   = lane & 15;
    const int hi   = lane >> 4;

    const _Float16* xrow = xb + (size_t)(i0 + lo) * F_INN;
    const _Float16* wb   = WbT + (size_t)h * F_OUTT * F_INN;

    v8f acc[4] = {v8f{}, v8f{}, v8f{}, v8f{}};

    for (int k = 0; k < F_INN; k += 32) {
        // A fragment: row = i0+lo ; elems 0-15: K=k+hi*16+{0..15}
        v16h afrag = *(const v16h*)(xrow + k + hi * 16);
        #pragma unroll
        for (int ot = 0; ot < 4; ++ot) {
            int c = ot * 16 + lo;   // B col
            v16h bfrag = *(const v16h*)(wb + (size_t)c * F_INN + k + hi * 16);
            acc[ot] = __builtin_amdgcn_wmma_f32_16x16x32_f16(
                false, afrag, false, bfrag, (short)0, acc[ot], false, false);
        }
    }
    // Epilogue: C elem v -> row i0 + v + 8*hi, col ot*16+lo. Store transposed (o-major).
    #pragma unroll
    for (int ot = 0; ot < 4; ++ot) {
        int o = ot * 16 + lo;
        v8h st;
        #pragma unroll
        for (int v = 0; v < 8; ++v) st[v] = (_Float16)acc[ot][v];
        *(v8h*)(WhT + (size_t)(h * F_OUTT + o) * N_NODES + i0 + hi * 8) = st;
    }
}

// ---------------- f1/f2 score vectors ----------------
__global__ void fvec_kernel(const _Float16* __restrict__ WhT, const float* __restrict__ a,
                            float* __restrict__ f1, float* __restrict__ f2) {
    int h = blockIdx.x >> 4;                       // grid = HEADS * 16
    int n = ((blockIdx.x & 15) << 8) + threadIdx.x;
    const _Float16* w  = WhT + (size_t)h * F_OUTT * N_NODES + n;
    const float*    ah = a + h * 2 * F_OUTT;
    float s1 = 0.f, s2 = 0.f;
    #pragma unroll 8
    for (int o = 0; o < F_OUTT; ++o) {
        float v = (float)w[(size_t)o * N_NODES];
        s1 += v * ah[o];
        s2 += v * ah[F_OUTT + o];
    }
    f1[h * N_NODES + n] = s1;
    f2[h * N_NODES + n] = s2;
}

// ---------------- softmax stats (row max + inverse sum), all heads fused per adj read --
__global__ void stats_kernel(const int* __restrict__ adj, const float* __restrict__ f1,
                             const float* __restrict__ f2,
                             float* __restrict__ mrow, float* __restrict__ linv) {
    const int i = blockIdx.x;
    const int t = threadIdx.x;
    __shared__ float red[256];
    const int* arow = adj + (size_t)i * N_NODES;

    float f1v[HEADS];
    #pragma unroll
    for (int h = 0; h < HEADS; ++h) f1v[h] = f1[h * N_NODES + i];

    float mloc[HEADS];
    #pragma unroll
    for (int h = 0; h < HEADS; ++h) mloc[h] = -3.0e38f;

    for (int j = t; j < N_NODES; j += 256) {
        if (arow[j] > 0) {
            #pragma unroll
            for (int h = 0; h < HEADS; ++h) {
                float e = f1v[h] + f2[h * N_NODES + j];
                e = e > 0.f ? e : ALPHAF * e;
                mloc[h] = fmaxf(mloc[h], e);
            }
        }
    }
    float mfin[HEADS];
    for (int h = 0; h < HEADS; ++h) {
        red[t] = mloc[h]; __syncthreads();
        for (int s = 128; s > 0; s >>= 1) {
            if (t < s) red[t] = fmaxf(red[t], red[t + s]);
            __syncthreads();
        }
        mfin[h] = red[0]; __syncthreads();
    }
    float sloc[HEADS];
    #pragma unroll
    for (int h = 0; h < HEADS; ++h) sloc[h] = 0.f;
    for (int j = t; j < N_NODES; j += 256) {
        if (arow[j] > 0) {
            #pragma unroll
            for (int h = 0; h < HEADS; ++h) {
                float e = f1v[h] + f2[h * N_NODES + j];
                e = e > 0.f ? e : ALPHAF * e;
                sloc[h] += __expf(e - mfin[h]);
            }
        }
    }
    for (int h = 0; h < HEADS; ++h) {
        red[t] = sloc[h]; __syncthreads();
        for (int s = 128; s > 0; s >>= 1) {
            if (t < s) red[t] += red[t + s];
            __syncthreads();
        }
        if (t == 0) {
            mrow[h * N_NODES + i] = mfin[h];
            linv[h * N_NODES + i] = 1.0f / red[0];
        }
        __syncthreads();
    }
}

// ---------------- fused masked-softmax aggregation: out = softmax(e) @ Wh -------------
// grid = N/16 blocks, 8 waves/block, wave = head. adj rows shared via cache.
__global__ void agg_kernel(const int* __restrict__ adj, const float* __restrict__ f1,
                           const float* __restrict__ f2, const float* __restrict__ mrow,
                           const float* __restrict__ linv, const _Float16* __restrict__ WhT,
                           float* __restrict__ out) {
    const int i0   = blockIdx.x * 16;
    const int h    = threadIdx.x >> 5;    // head = wave id
    const int lane = threadIdx.x & 31;
    const int lo   = lane & 15;
    const int hi   = lane >> 4;
    const int i    = i0 + lo;             // A-fragment row for this lane

    const float f1v = f1[h * N_NODES + i];
    const float mv  = mrow[h * N_NODES + i];
    const float* f2h = f2 + (size_t)h * N_NODES;
    const _Float16* wt = WhT + (size_t)h * F_OUTT * N_NODES;
    const int* arow = adj + (size_t)i * N_NODES;

    v8f acc[4] = {v8f{}, v8f{}, v8f{}, v8f{}};

    for (int j0 = 0; j0 < N_NODES; j0 += 32) {
        int jn = (j0 + 512 < N_NODES) ? j0 + 512 : j0;
        __builtin_prefetch(arow + jn, 0, 1);   // global_prefetch_b8 of next adj tile

        // adjacency + f2 for this lane's 16 K-slots
        v4i a0 = *(const v4i*)(arow + j0 + hi * 8);
        v4i a1 = *(const v4i*)(arow + j0 + hi * 8 + 4);
        v4i a2 = *(const v4i*)(arow + j0 + 16 + hi * 8);
        v4i a3 = *(const v4i*)(arow + j0 + 16 + hi * 8 + 4);
        v4f p0 = *(const v4f*)(f2h + j0 + hi * 8);
        v4f p1 = *(const v4f*)(f2h + j0 + hi * 8 + 4);
        v4f p2 = *(const v4f*)(f2h + j0 + 16 + hi * 8);
        v4f p3 = *(const v4f*)(f2h + j0 + 16 + hi * 8 + 4);

        v16h pfrag;
        #pragma unroll
        for (int e = 0; e < 4; ++e) {
            float s;
            s = f1v + p0[e]; s = s > 0.f ? s : ALPHAF * s;
            pfrag[e]      = (_Float16)((a0[e] > 0) ? __expf(s - mv) : 0.0f);
            s = f1v + p1[e]; s = s > 0.f ? s : ALPHAF * s;
            pfrag[4 + e]  = (_Float16)((a1[e] > 0) ? __expf(s - mv) : 0.0f);
            s = f1v + p2[e]; s = s > 0.f ? s : ALPHAF * s;
            pfrag[8 + e]  = (_Float16)((a2[e] > 0) ? __expf(s - mv) : 0.0f);
            s = f1v + p3[e]; s = s > 0.f ? s : ALPHAF * s;
            pfrag[12 + e] = (_Float16)((a3[e] > 0) ? __expf(s - mv) : 0.0f);
        }
        #pragma unroll
        for (int ot = 0; ot < 4; ++ot) {
            const _Float16* bp = wt + (size_t)(ot * 16 + lo) * N_NODES + j0 + hi * 8;
            v8h blo = *(const v8h*)(bp);
            v8h bhi = *(const v8h*)(bp + 16);
            v16h bfrag = __builtin_shufflevector(blo, bhi, 0,1,2,3,4,5,6,7,8,9,10,11,12,13,14,15);
            acc[ot] = __builtin_amdgcn_wmma_f32_16x16x32_f16(
                false, pfrag, false, bfrag, (short)0, acc[ot], false, false);
        }
    }

    // normalize by row sums + write concat-head output [N, H*F_OUT]
    float lv[8];
    #pragma unroll
    for (int v = 0; v < 8; ++v) lv[v] = linv[h * N_NODES + i0 + hi * 8 + v];
    #pragma unroll
    for (int ot = 0; ot < 4; ++ot) {
        int col = h * F_OUTT + ot * 16 + lo;
        #pragma unroll
        for (int v = 0; v < 8; ++v) {
            out[(size_t)(i0 + hi * 8 + v) * (HEADS * F_OUTT) + col] = acc[ot][v] * lv[v];
        }
    }
}

// ---------------- host launcher ----------------
extern "C" void kernel_launch(void* const* d_in, const int* in_sizes, int n_in,
                              void* d_out, int out_size, void* d_ws, size_t ws_size,
                              hipStream_t stream) {
    const float* x   = (const float*)d_in[0];
    const int*   adj = (const int*)d_in[1];
    const float* W   = (const float*)d_in[2];
    const float* a   = (const float*)d_in[3];
    float* out = (float*)d_out;

    char* ws = (char*)d_ws;
    size_t off = 0;
    _Float16* xb  = (_Float16*)(ws + off); off += (size_t)N_NODES * F_INN * sizeof(_Float16);
    _Float16* WbT = (_Float16*)(ws + off); off += (size_t)HEADS * F_OUTT * F_INN * sizeof(_Float16);
    _Float16* WhT = (_Float16*)(ws + off); off += (size_t)HEADS * F_OUTT * N_NODES * sizeof(_Float16);
    float* f1   = (float*)(ws + off); off += (size_t)HEADS * N_NODES * sizeof(float);
    float* f2   = (float*)(ws + off); off += (size_t)HEADS * N_NODES * sizeof(float);
    float* mrow = (float*)(ws + off); off += (size_t)HEADS * N_NODES * sizeof(float);
    float* linv = (float*)(ws + off); off += (size_t)HEADS * N_NODES * sizeof(float);

    cvt_x_kernel<<<(N_NODES * F_INN) / 256, 256, 0, stream>>>(x, xb);
    cvt_w_kernel<<<(HEADS * F_INN * F_OUTT) / 256, 256, 0, stream>>>(W, WbT);
    wh_gemm_kernel<<<(N_NODES / 16) * HEADS, 32, 0, stream>>>(xb, WbT, WhT);
    fvec_kernel<<<HEADS * (N_NODES / 256), 256, 0, stream>>>(WhT, a, f1, f2);
    stats_kernel<<<N_NODES, 256, 0, stream>>>(adj, f1, f2, mrow, linv);
    agg_kernel<<<N_NODES / 16, 256, 0, stream>>>(adj, f1, f2, mrow, linv, WhT, out);
}
